// Transformer_75703093559816
// MI455X (gfx1250) — matrix operand order repaired and checked by reading, not compiled, not run on hardware
//
#include <hip/hip_runtime.h>
#include <hip/hip_bf16.h>

// Model constants (match reference)
#define BB   4
#define SSL  512
#define DDM  512
#define HHN  8
#define HDE  64
#define LLY  6
#define VVC  32000
#define DFF  1024

typedef __attribute__((ext_vector_type(16))) _Float16 v16h;
typedef __attribute__((ext_vector_type(8)))  _Float16 v8h;
typedef __attribute__((ext_vector_type(8)))  float    v8f;

// ---------------------------------------------------------------------------
// Embedding + sinusoidal positional encoding (dual fp32 + f16 output):
// out[b,s,d] = E[tok[b,s], d] * sqrt(D) + pe(s, d)
// ---------------------------------------------------------------------------
__global__ void embed_pe_kernel(const int* __restrict__ tok,
                                const float* __restrict__ E,
                                float* __restrict__ out,
                                _Float16* __restrict__ outh, int BS)
{
    int idx = blockIdx.x * blockDim.x + threadIdx.x;
    if (idx >= BS * DDM) return;
    int d  = idx & (DDM - 1);
    int bs = idx >> 9;               // DDM == 512
    int s  = bs & (SSL - 1);         // SSL == 512
    int t  = tok[bs];
    int j  = d >> 1;
    float ang = (float)s * __expf(-(float)j * (9.210340371976184f / 128.0f));
    float pe  = (d & 1) ? __cosf(ang) : __sinf(ang);
    float r = E[(long long)t * DDM + d] * 22.62741699796952f + pe;
    out[idx]  = r;
    outh[idx] = (_Float16)r;
}

// ---------------------------------------------------------------------------
// fp32 -> f16 elementwise convert (for E)
// ---------------------------------------------------------------------------
__global__ void convert_f16_kernel(const float* __restrict__ in,
                                   _Float16* __restrict__ out, long long n)
{
    long long i = (long long)blockIdx.x * blockDim.x + threadIdx.x;
    if (i < n) out[i] = (_Float16)in[i];
}

// ---------------------------------------------------------------------------
// fp32 [R x C] row-major -> f16 [C x R] row-major (batched over z)
// ---------------------------------------------------------------------------
__global__ void transpose_f16_kernel(const float* __restrict__ in,
                                     _Float16* __restrict__ out, int R, int C)
{
    long long base = (long long)blockIdx.z * R * C;
    int idx = blockIdx.x * blockDim.x + threadIdx.x;
    if (idx >= R * C) return;
    int r = idx / C, c = idx - r * C;
    out[base + (long long)c * R + r] = (_Float16)in[base + idx];
}

// ---------------------------------------------------------------------------
// Fused residual-add + LayerNorm over rows of 512 (one wave32 per row).
// Writes fp32 (residual chain) and f16 (next GEMM input).
// ---------------------------------------------------------------------------
__global__ void add_ln_kernel(const float* __restrict__ a,
                              const float* __restrict__ bres,
                              const float* __restrict__ g,
                              const float* __restrict__ be,
                              float* __restrict__ out,
                              _Float16* __restrict__ outh, int rows)
{
    int row = blockIdx.x * (blockDim.x >> 5) + (threadIdx.x >> 5);
    if (row >= rows) return;
    int lane = threadIdx.x & 31;
    const float* ar = a    + (long long)row * DDM;
    const float* br = bres + (long long)row * DDM;
    float vals[16];
    float s = 0.f;
#pragma unroll
    for (int j = 0; j < 16; ++j) {
        int i = lane + 32 * j;
        vals[j] = ar[i] + br[i];
        s += vals[j];
    }
#pragma unroll
    for (int off = 16; off > 0; off >>= 1) s += __shfl_xor(s, off, 32);
    float mean = s * (1.0f / DDM);
    float v2 = 0.f;
#pragma unroll
    for (int j = 0; j < 16; ++j) { float d = vals[j] - mean; v2 += d * d; }
#pragma unroll
    for (int off = 16; off > 0; off >>= 1) v2 += __shfl_xor(v2, off, 32);
    float inv = rsqrtf(v2 * (1.0f / DDM) + 1e-5f);
    float* orow = out + (long long)row * DDM;
    _Float16* ohrow = outh + (long long)row * DDM;
#pragma unroll
    for (int j = 0; j < 16; ++j) {
        int i = lane + 32 * j;
        float r = g[i] * (vals[j] - mean) * inv + be[i];
        orow[i]  = r;
        ohrow[i] = (_Float16)r;
    }
}

// ---------------------------------------------------------------------------
// Masked softmax over f16 score rows of 512 (one wave32 per (b,h,q) row).
// fp32 accumulation; key-padding + optional causal mask as -1e9.
// ---------------------------------------------------------------------------
__global__ void softmax_mask_f16_kernel(_Float16* __restrict__ sc,
                                        const int* __restrict__ mask,
                                        int causal)
{
    int row = blockIdx.x * (blockDim.x >> 5) + (threadIdx.x >> 5);
    if (row >= BB * HHN * SSL) return;
    int lane = threadIdx.x & 31;
    int qi = row & (SSL - 1);
    int b  = row / (HHN * SSL);
    _Float16* p = sc + (long long)row * SSL;
    const int* mrow = mask + (long long)b * SSL;
    float vals[16];
    float mx = -3.4e38f;
#pragma unroll
    for (int j = 0; j < 16; ++j) {
        int kc = lane + 32 * j;
        float v = (float)p[kc];
        bool ok = (mrow[kc] != 0) && (!causal || (kc <= qi));
        v = ok ? v : -1e9f;
        vals[j] = v;
        mx = fmaxf(mx, v);
    }
#pragma unroll
    for (int off = 16; off > 0; off >>= 1) mx = fmaxf(mx, __shfl_xor(mx, off, 32));
    float sum = 0.f;
#pragma unroll
    for (int j = 0; j < 16; ++j) { vals[j] = __expf(vals[j] - mx); sum += vals[j]; }
#pragma unroll
    for (int off = 16; off > 0; off >>= 1) sum += __shfl_xor(sum, off, 32);
    float inv = 1.0f / sum;
#pragma unroll
    for (int j = 0; j < 16; ++j) p[lane + 32 * j] = (_Float16)(vals[j] * inv);
}

// ---------------------------------------------------------------------------
// Batched f16 WMMA GEMM, fp32 accumulate:  C = relu?( alpha * A@B' + bias )
// A:  f16 [M x K] row-major (lda)
// Bm: f16 [N x K] row-major (ldb)  ("TRANSB" layout, contiguous K per lane)
// CMODE 0: fp32 [M x N] row-major (ldc = row stride)
// CMODE 1: f16  [M x N] row-major (ldc = row stride)
// CMODE 2: f16  [N x M] transposed store (ldc = column stride; one packed
//          16B v8h store per tile since the 8 acc rows are M-contiguous)
// Per-z offsets: ptr += (z/zdiv)*hi + (z%zdiv)*lo  (element units).
// Wave tile: (16*MT) x (16*NT); B fragments shared across MT A fragments
// (MT*NT WMMAs per K=32 step from MT*2 + NT*2 b128 loads).
// Requires M%(16*MT)==0, N%(16*NT)==0, K%32==0 -> EXEC all-ones for WMMA.
// ---------------------------------------------------------------------------
template<int CMODE, int MT, int NT>
__global__ void gemm_f16_wmma_kernel(
    const _Float16* __restrict__ A, int lda, long long Ahi, long long Alo,
    const _Float16* __restrict__ Bm, int ldb, long long Bhi, long long Blo,
    const float* __restrict__ bias, long long bhi, long long blo,
    void* __restrict__ Cv, int ldc, long long Chi, long long Clo,
    int M, int N, int K, int zdiv, float alpha, int relu)
{
    int z  = blockIdx.z;
    int zh = z / zdiv, zl = z - zh * zdiv;
    A  += (long long)zh * Ahi + (long long)zl * Alo;
    Bm += (long long)zh * Bhi + (long long)zl * Blo;
    long long coff = (long long)zh * Chi + (long long)zl * Clo;
    if (bias) bias += (long long)zh * bhi + (long long)zl * blo;

    const int Mg = M / (16 * MT);
    const int Ng = N / (16 * NT);
    int wid = blockIdx.x * (blockDim.x >> 5) + (threadIdx.x >> 5);
    if (wid >= Mg * Ng) return;           // wave-uniform exit; EXEC stays full
    int m0 = (wid % Mg) * (16 * MT);
    int n0 = (wid / Mg) * (16 * NT);

    int lane = threadIdx.x & 31;
    int half = lane >> 4;                  // 0: lanes 0-15, 1: lanes 16-31
    int l15  = lane & 15;

    v8f acc[MT][NT];
    const v8f vzero = {0.f, 0.f, 0.f, 0.f, 0.f, 0.f, 0.f, 0.f};
#pragma unroll
    for (int mt = 0; mt < MT; ++mt)
#pragma unroll
        for (int t = 0; t < NT; ++t) acc[mt][t] = vzero;

    // A fragment (ISA 16x32 f16 layout): lane row = m0 + mt*16 + l15;
    // K chunks: [8*half .. 8*half+7] and [16+8*half .. 16+8*half+7]
    const _Float16* Abase[MT];
#pragma unroll
    for (int mt = 0; mt < MT; ++mt)
        Abase[mt] = A + (long long)(m0 + mt * 16 + l15) * lda + 8 * half;

    for (int k = 0; k < K; k += 32) {
        v16h af[MT];
#pragma unroll
        for (int mt = 0; mt < MT; ++mt) {
            v8h alo = *(const v8h*)(Abase[mt] + k);
            v8h ahi = *(const v8h*)(Abase[mt] + k + 16);
            af[mt] = __builtin_shufflevector(alo, ahi,
                0, 1, 2, 3, 4, 5, 6, 7, 8, 9, 10, 11, 12, 13, 14, 15);
        }
#pragma unroll
        for (int t = 0; t < NT; ++t) {
            int ncol = n0 + t * 16 + l15;
            // B fragment: lane col = ncol; K chunks [8*half..+7] and [16+8*half..+7]
            const _Float16* Bp = Bm + (long long)ncol * ldb + k + 8 * half;
            v16h bf = __builtin_shufflevector(*(const v8h*)Bp, *(const v8h*)(Bp + 16),
                0, 1, 2, 3, 4, 5, 6, 7, 8, 9, 10, 11, 12, 13, 14, 15);
#pragma unroll
            for (int mt = 0; mt < MT; ++mt)
                acc[mt][t] = __builtin_amdgcn_wmma_f32_16x16x32_f16(
                    false, af[mt], false, bf, (short)0, acc[mt][t], false, false);
        }
    }

#pragma unroll
    for (int t = 0; t < NT; ++t) {
        int ncol = n0 + t * 16 + l15;
        float bv = bias ? bias[ncol] : 0.f;
#pragma unroll
        for (int mt = 0; mt < MT; ++mt) {
            int mb = m0 + mt * 16 + 8 * half;
            if (CMODE == 0) {
                float* C = (float*)Cv + coff + ncol;
#pragma unroll
                for (int j = 0; j < 8; ++j) {
                    float v = acc[mt][t][j] * alpha + bv;
                    if (relu) v = fmaxf(v, 0.f);
                    C[(long long)(mb + j) * ldc] = v;
                }
            } else if (CMODE == 1) {
                _Float16* C = (_Float16*)Cv + coff + ncol;
#pragma unroll
                for (int j = 0; j < 8; ++j) {
                    float v = acc[mt][t][j] * alpha + bv;
                    if (relu) v = fmaxf(v, 0.f);
                    C[(long long)(mb + j) * ldc] = (_Float16)v;
                }
            } else {
                _Float16* C = (_Float16*)Cv + coff;
                v8h hv;
#pragma unroll
                for (int j = 0; j < 8; ++j) {
                    float v = acc[mt][t][j] * alpha + bv;
                    if (relu) v = fmaxf(v, 0.f);
                    hv[j] = (_Float16)v;
                }
                *(v8h*)(C + (long long)ncol * ldc + mb) = hv;
            }
        }
    }
}

// ---------------------------------------------------------------------------
// Host-side helpers
// ---------------------------------------------------------------------------
static void gemm(hipStream_t st, int cmode,
    const _Float16* A, int lda, long long Ahi, long long Alo,
    const _Float16* Bm, int ldb, long long Bhi, long long Blo,
    const float* bias, long long bhi, long long blo,
    void* C, int ldc, long long Chi, long long Clo,
    int M, int N, int K, int nz, int zdiv, float alpha, int relu)
{
    int waves = (M >> 5) * (N >> 6);       // MT=2, NT=4 -> 32 x 64 per wave
    dim3 grid((waves + 3) / 4, 1, nz);
    dim3 block(128, 1, 1);                 // 4 waves per block
    if (cmode == 0)
        gemm_f16_wmma_kernel<0, 2, 4><<<grid, block, 0, st>>>(A, lda, Ahi, Alo,
            Bm, ldb, Bhi, Blo, bias, bhi, blo, C, ldc, Chi, Clo, M, N, K, zdiv, alpha, relu);
    else if (cmode == 1)
        gemm_f16_wmma_kernel<1, 2, 4><<<grid, block, 0, st>>>(A, lda, Ahi, Alo,
            Bm, ldb, Bhi, Blo, bias, bhi, blo, C, ldc, Chi, Clo, M, N, K, zdiv, alpha, relu);
    else
        gemm_f16_wmma_kernel<2, 2, 4><<<grid, block, 0, st>>>(A, lda, Ahi, Alo,
            Bm, ldb, Bhi, Blo, bias, bhi, blo, C, ldc, Chi, Clo, M, N, K, zdiv, alpha, relu);
}

static void launch_transpose(hipStream_t st, const float* in, _Float16* out,
                             int R, int C, int nz)
{
    int tot = R * C;
    dim3 grid((tot + 255) / 256, 1, nz);
    transpose_f16_kernel<<<grid, 256, 0, st>>>(in, out, R, C);
}

// Multi-head attention. W*T: per-layer slice of transposed f16 weights,
// laid out [H, HD, D]. Writes head-concatenated f16 context into ao [B,S,D].
static void run_mha(hipStream_t st,
    const _Float16* xqh, const _Float16* xkvh,
    const _Float16* WqT, const float* bq,
    const _Float16* WkT, const float* bk,
    const _Float16* WvT, const float* bv,
    const int* mask, int causal,
    _Float16* qb, _Float16* kb, _Float16* vT, _Float16* sc, _Float16* ao)
{
    const long long SD   = (long long)SSL * DDM;
    const long long SHD  = (long long)SSL * HDE;
    const long long HSHD = (long long)HHN * SHD;
    const long long SS2  = (long long)SSL * SSL;
    const long long HDD  = (long long)HDE * DDM;

    // q/k: per (b,h) [S,D] x [D,HD] + bias -> f16 row-major [S,HD]
    gemm(st, 1, xqh, DDM, SD, 0, WqT, DDM, 0, HDD, bq, 0, HDE,
         qb, HDE, HSHD, SHD, SSL, HDE, DDM, BB * HHN, HHN, 1.0f, 0);
    gemm(st, 1, xkvh, DDM, SD, 0, WkT, DDM, 0, HDD, bk, 0, HDE,
         kb, HDE, HSHD, SHD, SSL, HDE, DDM, BB * HHN, HHN, 1.0f, 0);
    // v: transposed store -> vT per (b,h) is [HD, S] f16
    gemm(st, 2, xkvh, DDM, SD, 0, WvT, DDM, 0, HDD, bv, 0, HDE,
         vT, SSL, HSHD, SHD, SSL, HDE, DDM, BB * HHN, HHN, 1.0f, 0);
    // scores = q @ k^T / sqrt(HD) -> f16 in-place softmax buffer
    gemm(st, 1, qb, HDE, HSHD, SHD, kb, HDE, HSHD, SHD, nullptr, 0, 0,
         sc, SSL, (long long)HHN * SS2, SS2,
         SSL, SSL, HDE, BB * HHN, HHN, 0.125f, 0);
    int rows = BB * HHN * SSL;
    softmax_mask_f16_kernel<<<(rows + 7) / 8, 256, 0, st>>>(sc, mask, causal);
    // context = probs @ v: B operand is vT [HD x S]; store head-concatenated
    gemm(st, 1, sc, SSL, (long long)HHN * SS2, SS2, vT, SSL, HSHD, SHD,
         nullptr, 0, 0, ao, DDM, SD, (long long)HDE,
         SSL, HDE, SSL, BB * HHN, HHN, 1.0f, 0);
}

// ---------------------------------------------------------------------------
// kernel_launch
// ---------------------------------------------------------------------------
extern "C" void kernel_launch(void* const* d_in, const int* in_sizes, int n_in,
                              void* d_out, int out_size, void* d_ws, size_t ws_size,
                              hipStream_t stream)
{
    (void)in_sizes; (void)n_in; (void)out_size; (void)ws_size;

    // ---- inputs: tokens/masks, then params tree-flattened (sorted keys) ----
    const int*   src   = (const int*)d_in[0];
    const int*   emask = (const int*)d_in[1];
    const int*   tgt   = (const int*)d_in[2];
    const int*   dmask = (const int*)d_in[3];
    const float* E     = (const float*)d_in[4];
    const float* dW1   = (const float*)d_in[5];
    const float* dW2   = (const float*)d_in[6];
    const float* db1   = (const float*)d_in[7];
    const float* db2   = (const float*)d_in[8];
    const float* cWk   = (const float*)d_in[9];
    const float* cWo   = (const float*)d_in[10];
    const float* cWq   = (const float*)d_in[11];
    const float* cWv   = (const float*)d_in[12];
    const float* cbk   = (const float*)d_in[13];
    const float* cbo   = (const float*)d_in[14];
    const float* cbq   = (const float*)d_in[15];
    const float* cbv   = (const float*)d_in[16];
    const float* dln1b = (const float*)d_in[17];
    const float* dln1g = (const float*)d_in[18];
    const float* dln2b = (const float*)d_in[19];
    const float* dln2g = (const float*)d_in[20];
    const float* dln3b = (const float*)d_in[21];
    const float* dln3g = (const float*)d_in[22];
    const float* sWk   = (const float*)d_in[23];
    const float* sWo   = (const float*)d_in[24];
    const float* sWq   = (const float*)d_in[25];
    const float* sWv   = (const float*)d_in[26];
    const float* sbk   = (const float*)d_in[27];
    const float* sbo   = (const float*)d_in[28];
    const float* sbq   = (const float*)d_in[29];
    const float* sbv   = (const float*)d_in[30];
    const float* eW1   = (const float*)d_in[31];
    const float* eW2   = (const float*)d_in[32];
    const float* eWk   = (const float*)d_in[33];
    const float* eWo   = (const float*)d_in[34];
    const float* eWq   = (const float*)d_in[35];
    const float* eWv   = (const float*)d_in[36];
    const float* eb1   = (const float*)d_in[37];
    const float* eb2   = (const float*)d_in[38];
    const float* ebk   = (const float*)d_in[39];
    const float* ebo   = (const float*)d_in[40];
    const float* ebq   = (const float*)d_in[41];
    const float* ebv   = (const float*)d_in[42];
    const float* eln1b = (const float*)d_in[43];
    const float* eln1g = (const float*)d_in[44];
    const float* eln2b = (const float*)d_in[45];
    const float* eln2g = (const float*)d_in[46];

    // ---- sizes -------------------------------------------------------------
    const long long BSD  = (long long)BB * SSL * DDM;        // 1,048,576
    const long long BHSS = (long long)BB * HHN * SSL * SSL;  // 8,388,608
    const long long QL   = (long long)HHN * HDE * DDM;       // per-layer QKV^T
    const long long DD2  = (long long)DDM * DDM;
    const long long DDF  = (long long)DDM * DFF;
    const int BS = BB * SSL;

    // ---- workspace carve-up ------------------------------------------------
    float* ws = (float*)d_ws;
    float* xw = ws;                 // fp32 encoder activations [B,S,D]
    float* yw = xw + BSD;           // fp32 decoder activations
    float* t1 = yw + BSD;           // fp32 sublayer output
    float* t2 = t1 + BSD;           // fp32 decoder intermediate
    _Float16* h = (_Float16*)(t2 + BSD);
    _Float16* xh  = h; h += BSD;    // f16 mirrors
    _Float16* yh  = h; h += BSD;
    _Float16* t2h = h; h += BSD;
    _Float16* qb  = h; h += BSD;    // [B,H,S,HD]
    _Float16* kb  = h; h += BSD;
    _Float16* vT  = h; h += BSD;    // [B,H,HD,S]
    _Float16* ao  = h; h += BSD;    // attn context [B,S,D]
    _Float16* fh  = h; h += 2 * BSD;// FFN hidden [B,S,2D]
    _Float16* sc  = h; h += BHSS;   // scores/probs [B,H,S,S]
    // f16 transposed weights
    _Float16* Eh    = h; h += (long long)VVC * DDM;
    _Float16* eWqT  = h; h += LLY * QL;
    _Float16* eWkT  = h; h += LLY * QL;
    _Float16* eWvT  = h; h += LLY * QL;
    _Float16* eWoT  = h; h += LLY * DD2;
    _Float16* eW1T  = h; h += LLY * DDF;
    _Float16* eW2T  = h; h += LLY * DDF;
    _Float16* sWqT  = h; h += LLY * QL;
    _Float16* sWkT  = h; h += LLY * QL;
    _Float16* sWvT  = h; h += LLY * QL;
    _Float16* sWoT  = h; h += LLY * DD2;
    _Float16* cWqT  = h; h += LLY * QL;
    _Float16* cWkT  = h; h += LLY * QL;
    _Float16* cWvT  = h; h += LLY * QL;
    _Float16* cWoT  = h; h += LLY * DD2;
    _Float16* dW1T  = h; h += LLY * DDF;
    _Float16* dW2T  = h; h += LLY * DDF;

    // ---- one-pass weight conversion (f32 -> f16, transposed) --------------
    {
        long long nE = (long long)VVC * DDM;
        convert_f16_kernel<<<(int)((nE + 255) / 256), 256, 0, stream>>>(E, Eh, nE);
        launch_transpose(stream, eWq, eWqT, DDM, HDE, LLY * HHN);
        launch_transpose(stream, eWk, eWkT, DDM, HDE, LLY * HHN);
        launch_transpose(stream, eWv, eWvT, DDM, HDE, LLY * HHN);
        launch_transpose(stream, eWo, eWoT, DDM, DDM, LLY);
        launch_transpose(stream, eW1, eW1T, DDM, DFF, LLY);
        launch_transpose(stream, eW2, eW2T, DFF, DDM, LLY);
        launch_transpose(stream, sWq, sWqT, DDM, HDE, LLY * HHN);
        launch_transpose(stream, sWk, sWkT, DDM, HDE, LLY * HHN);
        launch_transpose(stream, sWv, sWvT, DDM, HDE, LLY * HHN);
        launch_transpose(stream, sWo, sWoT, DDM, DDM, LLY);
        launch_transpose(stream, cWq, cWqT, DDM, HDE, LLY * HHN);
        launch_transpose(stream, cWk, cWkT, DDM, HDE, LLY * HHN);
        launch_transpose(stream, cWv, cWvT, DDM, HDE, LLY * HHN);
        launch_transpose(stream, cWo, cWoT, DDM, DDM, LLY);
        launch_transpose(stream, dW1, dW1T, DDM, DFF, LLY);
        launch_transpose(stream, dW2, dW2T, DFF, DDM, LLY);
    }

    // ---- embeddings -------------------------------------------------------
    {
        int tot = BS * DDM;
        embed_pe_kernel<<<(tot + 255) / 256, 256, 0, stream>>>(src, E, xw, xh, BS);
        embed_pe_kernel<<<(tot + 255) / 256, 256, 0, stream>>>(tgt, E, yw, yh, BS);
    }

    // ---- encoder ----------------------------------------------------------
    for (int l = 0; l < LLY; ++l) {
        run_mha(stream, xh, xh,
                eWqT + l * QL, ebq + l * HHN * HDE,
                eWkT + l * QL, ebk + l * HHN * HDE,
                eWvT + l * QL, ebv + l * HHN * HDE,
                emask, 0, qb, kb, vT, sc, ao);
        gemm(stream, 0, ao, DDM, 0, 0, eWoT + l * DD2, DDM, 0, 0,
             ebo + l * DDM, 0, 0, t1, DDM, 0, 0, BS, DDM, DDM, 1, 1, 1.0f, 0);
        add_ln_kernel<<<(BS + 7) / 8, 256, 0, stream>>>(
            t1, xw, eln1g + l * DDM, eln1b + l * DDM, xw, xh, BS);
        gemm(stream, 1, xh, DDM, 0, 0, eW1T + l * DDF, DDM, 0, 0,
             eb1 + l * DFF, 0, 0, fh, DFF, 0, 0, BS, DFF, DDM, 1, 1, 1.0f, 1);
        gemm(stream, 0, fh, DFF, 0, 0, eW2T + l * DDF, DFF, 0, 0,
             eb2 + l * DDM, 0, 0, t1, DDM, 0, 0, BS, DDM, DFF, 1, 1, 1.0f, 0);
        add_ln_kernel<<<(BS + 7) / 8, 256, 0, stream>>>(
            t1, xw, eln2g + l * DDM, eln2b + l * DDM, xw, xh, BS);
    }

    // ---- decoder ----------------------------------------------------------
    for (int l = 0; l < LLY; ++l) {
        // causal self-attention on y
        run_mha(stream, yh, yh,
                sWqT + l * QL, sbq + l * HHN * HDE,
                sWkT + l * QL, sbk + l * HHN * HDE,
                sWvT + l * QL, sbv + l * HHN * HDE,
                dmask, 1, qb, kb, vT, sc, ao);
        gemm(stream, 0, ao, DDM, 0, 0, sWoT + l * DD2, DDM, 0, 0,
             sbo + l * DDM, 0, 0, t1, DDM, 0, 0, BS, DDM, DDM, 1, 1, 1.0f, 0);
        add_ln_kernel<<<(BS + 7) / 8, 256, 0, stream>>>(
            t1, yw, dln1g + l * DDM, dln1b + l * DDM, t2, t2h, BS);
        // cross-attention: queries from t2, keys/values from encoder output
        run_mha(stream, t2h, xh,
                cWqT + l * QL, cbq + l * HHN * HDE,
                cWkT + l * QL, cbk + l * HHN * HDE,
                cWvT + l * QL, cbv + l * HHN * HDE,
                emask, 0, qb, kb, vT, sc, ao);
        gemm(stream, 0, ao, DDM, 0, 0, cWoT + l * DD2, DDM, 0, 0,
             cbo + l * DDM, 0, 0, t1, DDM, 0, 0, BS, DDM, DDM, 1, 1, 1.0f, 0);
        add_ln_kernel<<<(BS + 7) / 8, 256, 0, stream>>>(
            t1, t2, dln2g + l * DDM, dln2b + l * DDM, t2, t2h, BS);
        // FFN
        gemm(stream, 1, t2h, DDM, 0, 0, dW1T + l * DDF, DDM, 0, 0,
             db1 + l * DFF, 0, 0, fh, DFF, 0, 0, BS, DFF, DDM, 1, 1, 1.0f, 1);
        gemm(stream, 0, fh, DFF, 0, 0, dW2T + l * DDF, DFF, 0, 0,
             db2 + l * DDM, 0, 0, t1, DDM, 0, 0, BS, DDM, DFF, 1, 1, 1.0f, 0);
        add_ln_kernel<<<(BS + 7) / 8, 256, 0, stream>>>(
            t1, t2, dln3g + l * DDM, dln3b + l * DDM, yw, yh, BS);
    }

    // ---- logits = y @ E^T  ([2048 x 512] x [512 x 32000], E already [V x D])
    gemm(stream, 0, yh, DDM, 0, 0, Eh, DDM, 0, 0, nullptr, 0, 0,
         (float*)d_out, VVC, 0, 0, BS, VVC, DDM, 1, 1, 1.0f, 0);
}
